// Attention_12446815224655
// MI455X (gfx1250) — compile-verified, matrix operand order checked by reading, not run on hardware
//
#include <hip/hip_runtime.h>
#include <hip/hip_bf16.h>

typedef __attribute__((ext_vector_type(16))) _Float16 v16h;
typedef __attribute__((ext_vector_type(8)))  _Float16 v8h;
typedef __attribute__((ext_vector_type(8)))  float    v8f;

#define AS3 __attribute__((address_space(3)))

#define WMMA_F32_F16(a, b, c) \
  __builtin_amdgcn_wmma_f32_16x16x32_f16(false, (a), false, (b), (short)0, (c), false, false)

// Load a 16x32 f16 A/B^T fragment from a row-major matrix (row stride ld halves).
// Per CDNA5 ISA: lane L%16 holds row M=L%16; elements 0..7 hold
// K = k0 + 8*laneHi + {0..7}, elements 8..15 hold K = k0 + 16 + 8*laneHi + {0..7}.
__device__ __forceinline__ v16h load_frag(const _Float16* __restrict__ p, int ld, int k0) {
  const int lane = threadIdx.x & 31;
  const int row  = lane & 15;
  const int hi   = lane >> 4;
  const _Float16* q = p + (size_t)row * (size_t)ld + k0 + (hi << 3);
  v8h c0 = *(const v8h*)(q);
  v8h c1 = *(const v8h*)(q + 16);
  v16h r;
#pragma unroll
  for (int i = 0; i < 8; ++i) { r[i] = c0[i]; r[i + 8] = c1[i]; }
  return r;
}

// One 16-byte async gather element: LDS[lds_byte_addr] <= MEM[gsrc], ASYNCcnt.
__device__ __forceinline__ void async_copy_b128(unsigned lds_byte_addr,
                                                const _Float16* gsrc) {
  asm volatile("global_load_async_to_lds_b128 %0, %1, off"
               :: "v"(lds_byte_addr),
                  "v"((unsigned long long)(uintptr_t)gsrc)
               : "memory");
}
__device__ __forceinline__ void wait_asynccnt0() {
  asm volatile("s_wait_asynccnt 0x0" ::: "memory");
}

// ---------------------------------------------------------------- converts ---
__global__ void __launch_bounds__(256) cvt_f32_f16(const float* __restrict__ src,
                                                   _Float16* __restrict__ dst, int n) {
  int i = blockIdx.x * 256 + threadIdx.x;
  if (i < n) dst[i] = (_Float16)src[i];
}

// dst[c*R + r] = (f16) src[r*C + c]   (transpose R x C -> C x R)
__global__ void __launch_bounds__(256) transpose_cvt(const float* __restrict__ src,
                                                     _Float16* __restrict__ dst,
                                                     int R, int C) {
  int i = blockIdx.x * 256 + threadIdx.x;
  if (i < R * C) {
    int c = i / R;
    int r = i - c * R;
    dst[(size_t)c * R + r] = (_Float16)src[(size_t)r * C + c];
  }
}

// vT[((b*16+h)*64 + d)*1024 + n] = qkv[(b*1024+n)*3072 + 2048 + h*64 + d]
__global__ void __launch_bounds__(256) extract_vT(const _Float16* __restrict__ qkv,
                                                  _Float16* __restrict__ vT) {
  int i = blockIdx.x * 256 + threadIdx.x;   // 8*16*64*1024 = 8388608 threads
  int n  = i & 1023;
  int d  = (i >> 10) & 63;
  int bh = i >> 16;
  int h  = bh & 15;
  int b  = bh >> 4;
  vT[i] = qkv[((size_t)(b * 1024 + n)) * 3072 + 2048 + h * 64 + d];
}

// ------------------------------------------------------------------- GEMMs ---
// Block tile: 128(M) x 64(N); 8 waves each own a 16x64 slice and SHARE the
// 64x32-halves B panel, double-buffered in LDS via GLOBAL_LOAD_ASYNC_TO_LDS.
// Per K-step: 1 async panel fetch per block (4 KB), A fragments straight from
// global (rows private to each wave), B fragments re-read as ds_load_b128.
// All four B fragments are made live before the WMMAs so the ds_loads cluster
// and the four v_wmma issue back-to-back.

#define GEMM_BODY(STORE_STMT)                                                   \
  __shared__ __align__(16) _Float16 btile[2][64 * 32];                          \
  const int tid    = threadIdx.x;                                               \
  const int wv     = tid >> 5;                                                  \
  const int ntiles = N >> 6;                                                    \
  const int mt     = blockIdx.x / ntiles;                                       \
  const int nt     = blockIdx.x - mt * ntiles;                                  \
  const int lane   = tid & 31;                                                  \
  const int row    = lane & 15;                                                 \
  const int hi     = lane >> 4;                                                 \
  /* B-panel fill assignment: 256 lanes x 16B == 64 rows x 32 halves */         \
  const int brow   = tid >> 2;                                                  \
  const int bchunk = tid & 3;                                                   \
  const _Float16* bsrc = BT + (size_t)(nt * 64 + brow) * K + bchunk * 8;        \
  const unsigned lds_base =                                                     \
      (unsigned)(uintptr_t)(AS3 char*)(&btile[0][0]);                           \
  const unsigned lane_off = (unsigned)((brow * 32 + bchunk * 8) * 2);           \
  const _Float16* Abase = A + (size_t)(mt * 128 + wv * 16) * K;                 \
  v8f acc[4] = {};                                                              \
  async_copy_b128(lds_base + lane_off, bsrc);         /* prologue: buf0 */      \
  int it = 0;                                                                   \
  for (int k0 = 0; k0 < K; k0 += 32, ++it) {                                    \
    wait_asynccnt0();            /* own panel portion has landed     */         \
    __syncthreads();             /* everyone's portion landed; other */         \
                                 /* buffer fully consumed by all     */         \
    if (k0 + 32 < K)                                                            \
      async_copy_b128(lds_base + ((unsigned)((it + 1) & 1)) * 4096u + lane_off, \
                      bsrc + k0 + 32);                                          \
    if (k0 + 64 < K)                                                            \
      __builtin_prefetch((const void*)(Abase + (size_t)row * K + k0 + 64), 0, 3);\
    v16h a = load_frag(Abase, K, k0);                                           \
    const _Float16* bb = &btile[it & 1][0];                                     \
    v16h bf0 = load_frag(bb + 0 * 512, 32, 0);                                  \
    v16h bf1 = load_frag(bb + 1 * 512, 32, 0);                                  \
    v16h bf2 = load_frag(bb + 2 * 512, 32, 0);                                  \
    v16h bf3 = load_frag(bb + 3 * 512, 32, 0);                                  \
    acc[0] = WMMA_F32_F16(a, bf0, acc[0]);                                      \
    acc[1] = WMMA_F32_F16(a, bf1, acc[1]);                                      \
    acc[2] = WMMA_F32_F16(a, bf2, acc[2]);                                      \
    acc[3] = WMMA_F32_F16(a, bf3, acc[3]);                                      \
  }                                                                             \
  _Pragma("unroll")                                                             \
  for (int t = 0; t < 4; ++t) {                                                 \
    _Pragma("unroll")                                                           \
    for (int r = 0; r < 8; ++r) {                                               \
      int m = mt * 128 + wv * 16 + r + 8 * hi;                                  \
      int n = nt * 64 + t * 16 + row;                                           \
      STORE_STMT;                                                               \
    }                                                                           \
  }

__global__ void __launch_bounds__(256) gemm_f16(const _Float16* __restrict__ A,
                                                const _Float16* __restrict__ BT,
                                                _Float16* __restrict__ C,
                                                int M, int N, int K) {
  GEMM_BODY(C[(size_t)m * N + n] = (_Float16)acc[t][r])
}

__global__ void __launch_bounds__(256) gemm_f32_bias(const _Float16* __restrict__ A,
                                                     const _Float16* __restrict__ BT,
                                                     const float* __restrict__ bias,
                                                     float* __restrict__ C,
                                                     int M, int N, int K) {
  GEMM_BODY(C[(size_t)m * N + n] = acc[t][r] + bias[n])
}

// -------------------------------------------------------- flash attention ---
// One wave handles 16 query rows of one (b, h).  Online softmax over 32-key
// tiles; P re-fragmented C-layout -> A-layout through 1 KB of LDS per wave.
__global__ void __launch_bounds__(256) attn_kernel(const _Float16* __restrict__ qkv,
                                                   const _Float16* __restrict__ vT,
                                                   _Float16* __restrict__ out) {
  __shared__ __align__(16) _Float16 plds[8][16 * 32];
  const int wlocal = threadIdx.x >> 5;
  const int wave   = blockIdx.x * 8 + wlocal;
  const int qt = wave & 63;
  const int h  = (wave >> 6) & 15;
  const int b  = wave >> 10;
  const int lane = threadIdx.x & 31;
  const int row  = lane & 15;
  const int hi   = lane >> 4;
  const float SCALE = 0.125f;   // 64^-0.5

  const size_t rowbase = (size_t)b * 1024 + (size_t)qt * 16;
  const _Float16* qptr = qkv + rowbase * 3072 + h * 64;
  const v16h a0 = load_frag(qptr, 3072, 0);    // d = 0..31
  const v16h a1 = load_frag(qptr, 3072, 32);   // d = 32..63

  v8f o[4] = {};
  float mr[8], lr[8];
#pragma unroll
  for (int r = 0; r < 8; ++r) { mr[r] = -__builtin_inff(); lr[r] = 0.0f; }

  const _Float16* kbase = qkv + (size_t)b * 1024 * 3072 + 1024 + h * 64;
  const _Float16* vbase = vT + ((size_t)(b * 16 + h) * 64) * 1024;
  _Float16* pw = &plds[wlocal][0];

  for (int kt = 0; kt < 1024; kt += 32) {
    const _Float16* k0p = kbase + (size_t)kt * 3072;
    const _Float16* k1p = kbase + (size_t)(kt + 16) * 3072;
    const v8f z = {};
    // all four K fragments live before the S WMMAs -> clustered loads
    v16h kb00 = load_frag(k0p, 3072, 0);
    v16h kb01 = load_frag(k0p, 3072, 32);
    v16h kb10 = load_frag(k1p, 3072, 0);
    v16h kb11 = load_frag(k1p, 3072, 32);
    v8f s0 = WMMA_F32_F16(a0, kb00, z);
    v8f s1 = WMMA_F32_F16(a0, kb10, z);
    s0 = WMMA_F32_F16(a1, kb01, s0);
    s1 = WMMA_F32_F16(a1, kb11, s1);

    float corr[8];
#pragma unroll
    for (int r = 0; r < 8; ++r) {
      float x0 = s0[r] * SCALE;
      float x1 = s1[r] * SCALE;
      // row max across the 16 lanes of this half-wave (xor masks 1..8 stay
      // inside a 16-lane group on wave32)
      float tmax = fmaxf(x0, x1);
#pragma unroll
      for (int msk = 1; msk <= 8; msk <<= 1)
        tmax = fmaxf(tmax, __shfl_xor(tmax, msk, 32));
      float mnew = fmaxf(mr[r], tmax);
      float c  = __expf(mr[r] - mnew);
      float p0 = __expf(x0 - mnew);
      float p1 = __expf(x1 - mnew);
      float rs = p0 + p1;
#pragma unroll
      for (int msk = 1; msk <= 8; msk <<= 1)
        rs += __shfl_xor(rs, msk, 32);
      lr[r] = lr[r] * c + rs;
      mr[r] = mnew;
      corr[r] = c;
      // stage P tile (16q x 32k) row-major in LDS for A-layout reload
      int m = r + 8 * hi;
      pw[m * 32 + row]      = (_Float16)p0;
      pw[m * 32 + 16 + row] = (_Float16)p1;
    }
#pragma unroll
    for (int t = 0; t < 4; ++t)
#pragma unroll
      for (int r = 0; r < 8; ++r) o[t][r] *= corr[r];

    __builtin_amdgcn_wave_barrier();          // wave-synchronous LDS handoff
    v16h pf = load_frag(pw, 32, 0);
    __builtin_amdgcn_wave_barrier();

    // all four V fragments live before the PV WMMAs
    v16h vf0 = load_frag(vbase + (size_t)(0 * 16) * 1024 + kt, 1024, 0);
    v16h vf1 = load_frag(vbase + (size_t)(1 * 16) * 1024 + kt, 1024, 0);
    v16h vf2 = load_frag(vbase + (size_t)(2 * 16) * 1024 + kt, 1024, 0);
    v16h vf3 = load_frag(vbase + (size_t)(3 * 16) * 1024 + kt, 1024, 0);
    o[0] = WMMA_F32_F16(pf, vf0, o[0]);
    o[1] = WMMA_F32_F16(pf, vf1, o[1]);
    o[2] = WMMA_F32_F16(pf, vf2, o[2]);
    o[3] = WMMA_F32_F16(pf, vf3, o[3]);
  }

  float inv[8];
#pragma unroll
  for (int r = 0; r < 8; ++r) inv[r] = 1.0f / lr[r];
#pragma unroll
  for (int t = 0; t < 4; ++t)
#pragma unroll
    for (int r = 0; r < 8; ++r) {
      size_t m = rowbase + r + 8 * hi;
      int n = h * 64 + t * 16 + row;
      out[m * 1024 + n] = (_Float16)(o[t][r] * inv[r]);
    }
}

// ------------------------------------------------------------------ launch ---
extern "C" void kernel_launch(void* const* d_in, const int* in_sizes, int n_in,
                              void* d_out, int out_size, void* d_ws, size_t ws_size,
                              hipStream_t stream) {
  const float* x     = (const float*)d_in[0];   // (8,1024,1024)
  const float* w_qkv = (const float*)d_in[1];   // (1024,3072)
  const float* w_out = (const float*)d_in[2];   // (1024,1024)
  const float* b_out = (const float*)d_in[3];   // (1024,)
  float* out = (float*)d_out;                   // (8,1024,1024)

  char* ws = (char*)d_ws;
  _Float16* x_h    = (_Float16*)(ws);                        // 16 MB
  _Float16* wqkvT  = (_Float16*)(ws + (16ull << 20));        //  6 MB
  _Float16* woutT  = (_Float16*)(ws + (22ull << 20));        //  2 MB
  _Float16* qkv_h  = (_Float16*)(ws + (24ull << 20));        // 48 MB
  _Float16* vT     = (_Float16*)(ws + (72ull << 20));        // 16 MB -> 88 MB total
  _Float16* attn_h = x_h;  // x dead after QKV GEMM; reuse region

  // 1) stage operands in f16
  cvt_f32_f16<<<8388608 / 256, 256, 0, stream>>>(x, x_h, 8388608);
  transpose_cvt<<<(1024 * 3072) / 256, 256, 0, stream>>>(w_qkv, wqkvT, 1024, 3072);
  transpose_cvt<<<(1024 * 1024) / 256, 256, 0, stream>>>(w_out, woutT, 1024, 1024);

  // 2) qkv = x @ w_qkv : block tile 128x64 -> (8192/128)*(3072/64) = 3072 blocks
  gemm_f16<<<3072, 256, 0, stream>>>(x_h, wqkvT, qkv_h, 8192, 3072, 1024);

  // 3) V^T per (b,h)
  extract_vT<<<8388608 / 256, 256, 0, stream>>>(qkv_h, vT);

  // 4) flash attention: 8192 waves / 8 = 1024 blocks
  attn_kernel<<<1024, 256, 0, stream>>>(qkv_h, vT, attn_h);

  // 5) out = attn @ w_out + b : (8192/128)*(1024/64) = 1024 blocks
  gemm_f32_bias<<<1024, 256, 0, stream>>>(attn_h, woutT, b_out, out, 8192, 1024, 1024);
}
